// MPQ_48730698940485
// MI455X (gfx1250) — matrix operand order repaired and checked by reading, not compiled, not run on hardware
//
#include <hip/hip_runtime.h>

typedef __attribute__((ext_vector_type(16))) _Float16 v16h;
typedef __attribute__((ext_vector_type(8)))  _Float16 v8h;
typedef __attribute__((ext_vector_type(8)))  float    v8f;

#define D_VEC 128
#define NSUB  8
#define DSUB  16
#define KCLUS 256
#define NCB   64

// ---------------- WMMA fragment loaders (this file's contiguous k order: k = 16*(lane/16) + elem) ----
// All loaders arrange contiguous 16B/32B LDS reads per lane -> ds_load_b128.

// A (16 pts x 32 K), base = [16][ldm] f16 row-major (row = point).
// lanes 0-15: M=lane, K = k0 + {0..15}; lanes 16-31: M=lane-16, K = k0 + {16..31}
static __device__ __forceinline__ v16h lds_a_frag(const _Float16* __restrict__ base,
                                                  int ldm, int lane, int k0) {
  const int row = lane & 15;
  const int ho  = (lane & 16) ? 8 : 0;
  const v8h lo = *(const v8h*)(base + row * ldm + k0 + 2 * ho);      // K = k0+2ho .. +7
  const v8h hi = *(const v8h*)(base + row * ldm + k0 + 2 * ho + 8);  // K = k0+2ho+8 .. +7
  return __builtin_shufflevector(lo, hi, 0, 1, 2, 3, 4, 5, 6, 7,
                                 8, 9, 10, 11, 12, 13, 14, 15);
}

// B (32 K x 16 N), baseT = [N][ldk] f16 *column-major tile* (row = N/cluster).
// lane n holds K = kb..kb+15 in order (2 per VGPR) -> one contiguous 32B row chunk.
static __device__ __forceinline__ v16h lds_b_frag(const _Float16* __restrict__ baseT,
                                                  int ldk, int lane, int k0, int n0) {
  const int n  = n0 + (lane & 15);
  const int kb = (lane & 16) ? 16 : 0;
  return *(const v16h*)(baseT + n * ldk + k0 + kb);
}

// A with real K=16 zero-padded to 32: lanes 16-31 (K=16..31) all zero.
static __device__ __forceinline__ v16h lds_a_frag16(const _Float16* __restrict__ base,
                                                    int ldm, int lane) {
  const int row = lane & 15;
  v8h z;
#pragma unroll
  for (int i = 0; i < 8; ++i) z[i] = (_Float16)0.f;
  const v8h lo = (lane & 16) ? z : *(const v8h*)(base + row * ldm);      // dims 0..7
  const v8h hi = (lane & 16) ? z : *(const v8h*)(base + row * ldm + 8);  // dims 8..15
  return __builtin_shufflevector(lo, hi, 0, 1, 2, 3, 4, 5, 6, 7,
                                 8, 9, 10, 11, 12, 13, 14, 15);
}

// B with real K=16 zero-padded to 32: lanes 16-31 (K=16..31) all zero.
// baseT = [256][16] f16 cluster-major; lane n reads one full 32B row.
static __device__ __forceinline__ v16h lds_b_frag16(const _Float16* __restrict__ baseT,
                                                    int lane, int n0) {
  v16h b;
  if (lane & 16) {
#pragma unroll
    for (int i = 0; i < 16; ++i) b[i] = (_Float16)0.f;
  } else {
    b = *(const v16h*)(baseT + (n0 + (lane & 15)) * DSUB);
  }
  return b;
}

// ---------------- kernel 0: workspace init (runs every call; graph-safe) -----

__global__ void pq_init_kernel(int* __restrict__ counts, int* __restrict__ cursor,
                               int* __restrict__ ntile, int* __restrict__ sidx, int sn) {
  const int gid = blockIdx.x * blockDim.x + threadIdx.x;
  if (gid < NCB) { counts[gid] = 0; cursor[gid] = 0; }
  if (gid == 0) ntile[0] = 0;
  for (int i = gid; i < sn; i += gridDim.x * blockDim.x) sidx[i] = -1;
}

// ---------------- kernel 1: selector labels via WMMA -------------------------
// block = 128 (4 waves); each wave owns a 16-point tile; 64 points / block.

#define W1 4
__global__ __launch_bounds__(128) void pq_labels_kernel(
    const float* __restrict__ x, const float* __restrict__ sc,
    int* __restrict__ labels, int* __restrict__ counts, int N) {
  __shared__ _Float16 cT[NCB][D_VEC];       // 16 KB, centroid-major (B-operand layout)
  __shared__ float    cnorm[NCB];
  __shared__ _Float16 xt[W1][16][D_VEC];    // 16 KB, point-major (A-operand layout)
  __shared__ float    xnorm[W1][16];
  __shared__ float    candv[W1][8][32];
  __shared__ int      candi[W1][8][32];

  const int tid  = threadIdx.x;
  const int lane = tid & 31;
  const int w    = tid >> 5;

  // stage selector centroids transposed: cT[l][d] = sc[d*64 + l] (coalesced reads)
  for (int idx = tid; idx < D_VEC * NCB; idx += 128) {
    const int d = idx >> 6, l = idx & 63;
    cT[l][d] = (_Float16)sc[idx];
  }
  __syncthreads();
  if (tid < NCB) {
    float s = 0.f;
    for (int d = 0; d < D_VEC; ++d) { float v = (float)cT[tid][d]; s += v * v; }
    cnorm[tid] = s;
  }

  // stage this wave's 16 points: xt[pt][d] = x[d, p0+pt]
  const int p0 = blockIdx.x * (16 * W1) + w * 16;
#pragma unroll 4
  for (int i = 0; i < 64; ++i) {
    const int idx = i * 32 + lane;
    const int pt = idx & 15, d = idx >> 4;
    xt[w][pt][d] = (_Float16)x[(size_t)d * N + p0 + pt];
  }
  __syncthreads();
  if (lane < 16) {
    float s = 0.f;
    for (int d = 0; d < D_VEC; ++d) { float v = (float)xt[w][lane][d]; s += v * v; }
    xnorm[w][lane] = s;
  }
  __syncthreads();

  // cross[pts, 64 centroids] = X(16x128) * C(128x64): 4 k-steps x 4 n-tiles
  v8f zero = {};
  v8f acc[4] = {zero, zero, zero, zero};
#pragma unroll
  for (int kk = 0; kk < 4; ++kk) {
    v16h a = lds_a_frag(&xt[w][0][0], D_VEC, lane, kk * 32);
#pragma unroll
    for (int nt = 0; nt < 4; ++nt) {
      v16h b = lds_b_frag(&cT[0][0], D_VEC, lane, kk * 32, nt * 16);
      acc[nt] = __builtin_amdgcn_wmma_f32_16x16x32_f16(false, a, false, b,
                                                       (short)0, acc[nt], false, false);
    }
  }

  // per-(VGPR,lane) running argmin over the 4 centroid tiles
  float bv[8]; int bi[8];
#pragma unroll
  for (int v = 0; v < 8; ++v) { bv[v] = 3.0e38f; bi[v] = 0; }
#pragma unroll
  for (int nt = 0; nt < 4; ++nt) {
#pragma unroll
    for (int v = 0; v < 8; ++v) {
      const int M   = v + ((lane & 16) ? 8 : 0);
      const int col = nt * 16 + (lane & 15);
      const float dist = xnorm[w][M] - 2.f * acc[nt][v] + cnorm[col];
      if (dist < bv[v]) { bv[v] = dist; bi[v] = col; }
    }
  }
#pragma unroll
  for (int v = 0; v < 8; ++v) { candv[w][v][lane] = bv[v]; candi[w][v][lane] = bi[v]; }
  __syncthreads();

  if (lane < 16) {               // point 'lane' lives at VGPR lane&7, lane half by lane<8
    const int vv = lane & 7, lo = (lane < 8) ? 0 : 16;
    float best = 3.0e38f; int lab = 0;
    for (int j = 0; j < 16; ++j) {
      const float vl = candv[w][vv][lo + j];
      if (vl < best) { best = vl; lab = candi[w][vv][lo + j]; }
    }
    labels[p0 + lane] = lab;
    atomicAdd(&counts[lab], 1);
  }
}

// ---------------- kernel 2: padded prefix sum + tile->label map --------------

__global__ __launch_bounds__(256) void pq_scan_kernel(
    const int* __restrict__ counts, int* __restrict__ poff,
    int* __restrict__ ntile, int* __restrict__ tlabel, int maxTiles) {
  __shared__ int ntl;
  if (threadIdx.x == 0) {
    int run = 0;
    for (int l = 0; l < NCB; ++l) { poff[l] = run; run += ((counts[l] + 15) >> 4) << 4; }
    poff[NCB] = run;
    ntile[0] = run >> 4;
    ntl = run >> 4;
  }
  __syncthreads();
  for (int t = threadIdx.x; t < maxTiles; t += 256) {
    if (t < ntl) {
      const int p = t * 16;
      int lab = 0;
      for (int l = 0; l < NCB; ++l)
        if (p >= poff[l]) lab = l;   // last label whose padded region starts <= p
      tlabel[t] = lab;
    }
  }
}

// ---------------- kernel 3: scatter points into label-sorted tiles -----------

__global__ void pq_scatter_kernel(const int* __restrict__ labels, const int* __restrict__ poff,
                                  int* __restrict__ cursor, int* __restrict__ sidx, int N) {
  const int gid = blockIdx.x * blockDim.x + threadIdx.x;
  if (gid < N) {
    const int l = labels[gid];
    const int pos = poff[l] + atomicAdd(&cursor[l], 1);
    sidx[pos] = gid;
  }
}

// ---------------- kernel 4: PQ encode + decode via WMMA ----------------------
// one wave per 16-point tile (all same codebook label); loops 8 subvectors.

__global__ __launch_bounds__(32) void pq_encode_decode_kernel(
    const float* __restrict__ x, const float* __restrict__ codebook,
    const int* __restrict__ sorted_idx, const int* __restrict__ tile_label,
    const int* __restrict__ n_tiles_p, float* __restrict__ out, int N) {
  const int t = blockIdx.x;
  if (t >= *n_tiles_p) return;            // uniform -> EXEC all-ones afterwards
  const int lane = threadIdx.x;

  __shared__ int      pts[16];
  __shared__ _Float16 xt[16][D_VEC];      // 4 KB: gathered points, point-major
  __shared__ _Float16 cbT[KCLUS][DSUB];   // 8 KB: codebook slice, cluster-major
  __shared__ float    cnorm[KCLUS];
  __shared__ float    xnorm[16];
  __shared__ float    candv[8][32];
  __shared__ int      candi[8][32];

  if (lane < 16) pts[lane] = sorted_idx[t * 16 + lane];
  __syncthreads();

  // gather x columns for this tile (padding slots -> 0)
#pragma unroll 4
  for (int i = 0; i < 64; ++i) {
    const int idx = i * 32 + lane;
    const int pt = idx & 15, d = idx >> 4;
    const int n = pts[pt];
    xt[pt][d] = (n >= 0) ? (_Float16)x[(size_t)d * N + n] : (_Float16)0.f;
  }
  const int label = tile_label[t];
  const size_t cbbase0 = (size_t)label * NSUB * DSUB * KCLUS;
  __syncthreads();

  for (int s = 0; s < NSUB; ++s) {
    const float* cb = codebook + cbbase0 + (size_t)s * DSUB * KCLUS;  // [16][256]
    if (s + 1 < NSUB)
      __builtin_prefetch(cb + DSUB * KCLUS, 0, 3);                    // global_prefetch

    // stage codebook slice transposed: cbT[k][d] = cb[d*256 + k] (coalesced reads)
#pragma unroll 4
    for (int i = 0; i < 128; ++i) {
      const int idx = i * 32 + lane;
      const int d = idx >> 8, k = idx & 255;
      cbT[k][d] = (_Float16)cb[idx];
    }
    __syncthreads();

    // per-cluster norms (8 clusters / lane, contiguous reads) and point norms
#pragma unroll
    for (int j = 0; j < 8; ++j) {
      const int k = lane * 8 + j;
      float sum = 0.f;
      for (int d = 0; d < DSUB; ++d) { float v = (float)cbT[k][d]; sum += v * v; }
      cnorm[k] = sum;
    }
    if (lane < 16) {
      float sum = 0.f;
      for (int d = 0; d < DSUB; ++d) { float v = (float)xt[lane][s * 16 + d]; sum += v * v; }
      xnorm[lane] = sum;
    }
    __syncthreads();

    // cross[16 pts, 256 clusters]: 16 WMMAs, K=16 zero-padded to 32
    v16h a = lds_a_frag16(&xt[0][s * 16], D_VEC, lane);
    float bv[8]; int bi[8];
#pragma unroll
    for (int v = 0; v < 8; ++v) { bv[v] = 3.0e38f; bi[v] = 0; }
#pragma unroll 4
    for (int nt = 0; nt < 16; ++nt) {
      v16h b = lds_b_frag16(&cbT[0][0], lane, nt * 16);
      v8f c = {};
      c = __builtin_amdgcn_wmma_f32_16x16x32_f16(false, a, false, b,
                                                 (short)0, c, false, false);
#pragma unroll
      for (int v = 0; v < 8; ++v) {
        const int M   = v + ((lane & 16) ? 8 : 0);
        const int col = nt * 16 + (lane & 15);
        const float dist = xnorm[M] - 2.f * c[v] + cnorm[col];
        if (dist < bv[v]) { bv[v] = dist; bi[v] = col; }
      }
    }
#pragma unroll
    for (int v = 0; v < 8; ++v) { candv[v][lane] = bv[v]; candi[v][lane] = bi[v]; }
    __syncthreads();

    // per-point argmin over its 16 lane-candidates, then decode from f32 codebook
    if (lane < 16) {
      const int vv = lane & 7, lo = (lane < 8) ? 0 : 16;
      float best = 3.0e38f; int code = 0;
      for (int j = 0; j < 16; ++j) {
        const float vl = candv[vv][lo + j];
        if (vl < best) { best = vl; code = candi[vv][lo + j]; }
      }
      const int n = pts[lane];
      if (n >= 0) {
        for (int d = 0; d < DSUB; ++d)
          out[(size_t)(s * 16 + d) * N + n] = cb[d * KCLUS + code];  // exact f32 values
      }
    }
    __syncthreads();   // before cbT/cand reuse next subvector
  }
}

// ---------------- launch -----------------------------------------------------

extern "C" void kernel_launch(void* const* d_in, const int* in_sizes, int n_in,
                              void* d_out, int out_size, void* d_ws, size_t ws_size,
                              hipStream_t stream) {
  const float* x        = (const float*)d_in[0];  // [128, N]
  const float* codebook = (const float*)d_in[1];  // [64, 8, 16, 256]
  const float* sc       = (const float*)d_in[2];  // [128, 64]
  float* out = (float*)d_out;                     // [128, N]
  const int N = in_sizes[0] / D_VEC;              // 131072

  // workspace layout (ints)
  int* labels = (int*)d_ws;                 // N
  int* counts = labels + N;                 // 64
  int* poff   = counts + NCB;               // 65
  int* cursor = poff + (NCB + 1);           // 64
  int* ntile  = cursor + NCB;               // 1
  const int maxTiles = N / 16 + NCB;        // worst-case padded tile count
  int* tlabel = ntile + 1;                  // maxTiles
  int* sidx   = tlabel + maxTiles;          // N + 64*16
  const int sn = N + NCB * 16;

  pq_init_kernel<<<(sn + 255) / 256, 256, 0, stream>>>(counts, cursor, ntile, sidx, sn);
  pq_labels_kernel<<<N / (16 * W1), 128, 0, stream>>>(x, sc, labels, counts, N);
  pq_scan_kernel<<<1, 256, 0, stream>>>(counts, poff, ntile, tlabel, maxTiles);
  pq_scatter_kernel<<<(N + 255) / 256, 256, 0, stream>>>(labels, poff, cursor, sidx, N);
  pq_encode_decode_kernel<<<maxTiles, 32, 0, stream>>>(x, codebook, sidx, tlabel, ntile, out, N);
}
